// AttentionLayer_25202868093362
// MI455X (gfx1250) — matrix operand order repaired and checked by reading, not compiled, not run on hardware
//
#include <hip/hip_runtime.h>
#include <hip/hip_bf16.h>

// ---------------------------------------------------------------------------
// Fused QKV projection (split-fp16 WMMA, fp32 accumulate) + per-node 4x4
// cross-head attention for MI455X (gfx1250, wave32).
// ---------------------------------------------------------------------------

typedef _Float16 v16h __attribute__((ext_vector_type(16)));
typedef _Float16 v8h  __attribute__((ext_vector_type(8)));
typedef float    v8f  __attribute__((ext_vector_type(8)));

#define IN_SIZE        256
#define OUT_SIZE       256
#define NUM_HEADS      4
#define HEAD           64
#define NTILES         48            // 3 matrices * 16 column tiles of 16
#define KCHUNKS        8             // 256 / 32
#define TILES_PER_WAVE 12            // 48 tiles / 4 waves
#define QKV_STRIDE     784           // 768 + 16 pad -> conflict-free LDS halves
#define WS_WBYTES      (NTILES * KCHUNKS * 2048)   // 786432 bytes packed weights
// packed block (one 32x16 B tile, hi+lo):
//   [0,512)    hi halves j=0..7   lane-major 16B
//   [512,1024) hi halves j=8..15
//   [1024,2048) same for lo
// j = 2*v + delta maps to WMMA B VGPR v, half delta.

__device__ __forceinline__ void split1(float f, _Float16& hi, _Float16& lo) {
    hi = (_Float16)f;
    lo = (_Float16)(f - (float)hi);
}

// ---------------------------------------------------------------------------
// Pre-pack: weights -> per-lane WMMA B layout (hi/lo f16 planes), biases -> cat
// ---------------------------------------------------------------------------
__global__ void pack_weights_kernel(const float* __restrict__ Wq,
                                    const float* __restrict__ Wk,
                                    const float* __restrict__ Wv,
                                    const float* __restrict__ bq,
                                    const float* __restrict__ bk,
                                    const float* __restrict__ bv,
                                    unsigned char* __restrict__ wsW,
                                    float* __restrict__ wsB) {
    int tid = blockIdx.x * blockDim.x + threadIdx.x;

    if (tid < 3 * OUT_SIZE) {                      // 768 bias values, cat Q,K,V
        int m = tid >> 8;
        int i = tid & 255;
        const float* bsrc = (m == 0) ? bq : (m == 1) ? bk : bv;
        wsB[tid] = bsrc[i];
    }

    if (tid < 3 * 16 * 8 * 32 * 16) {              // 196608 packed elements
        int j    = tid & 15;                        // half index within lane
        int ld   = (tid >> 4) & 31;                 // lane
        int c    = (tid >> 9) & 7;                  // K-chunk
        int t    = (tid >> 12) & 15;                // N tile within matrix
        int m    = tid >> 16;                       // 0=Q 1=K 2=V
        const float* W = (m == 0) ? Wq : (m == 1) ? Wk : Wv;

        int k    = c * 32 + ((j & 8) ? 16 : 0) + ((ld & 16) ? 8 : 0) + (j & 7);   // B layout K index
        int ncol = t * 16 + (ld & 15);
        float w  = W[(size_t)k * OUT_SIZE + ncol];
        _Float16 hi, lo;
        split1(w, hi, lo);

        int b = (m * 16 + t) * 8 + c;               // block id 0..383
        size_t off = (size_t)b * 2048 + ((j & 8) ? 512 : 0) + ld * 16 + (j & 7) * 2;
        *(_Float16*)(wsW + off)        = hi;
        *(_Float16*)(wsW + off + 1024) = lo;
    }
}

// ---------------------------------------------------------------------------
// Main fused kernel: one workgroup (4 waves) per 16-row tile of nodes.
// ---------------------------------------------------------------------------
__global__ __launch_bounds__(128, 1)
void qkv_attn_kernel(const float* __restrict__ h,
                     const unsigned char* __restrict__ wsW,
                     const float* __restrict__ wsB,
                     float* __restrict__ out,
                     int n) {
    __shared__ float sQKV[16 * QKV_STRIDE];   // 16 nodes x (q|k|v) padded
    __shared__ float sS[16 * 16];             // 16 nodes x 4x4 logits

    const int tid  = threadIdx.x;
    const int lane = tid & 31;
    const int wave = tid >> 5;
    const int m0   = blockIdx.x * 16;

    // ---- GEMM phase: C[16x768] = split_f16(h_tile) x split_f16(Wcat) -------
    int arow = m0 + (lane & 15);
    if (arow >= n) arow = n - 1;               // safe duplicate for tail tiles
    const float* hrow = h + (size_t)arow * IN_SIZE;
    const int aoff = (lane & 16) ? 8 : 0;      // A layout: lanes 16-31 hold K+8

    v8f acc[TILES_PER_WAVE];
#pragma unroll
    for (int i = 0; i < TILES_PER_WAVE; ++i)
        acc[i] = (v8f){0.f, 0.f, 0.f, 0.f, 0.f, 0.f, 0.f, 0.f};

    for (int c = 0; c < KCHUNKS; ++c) {
        // Build A fragment (16x32 f16 layout) straight from global h.
        v16h ahi, alo;
        {
            const float* ap = hrow + c * 32 + aoff;
            float4 f0 = *(const float4*)(ap);
            float4 f1 = *(const float4*)(ap + 4);
            float4 f2 = *(const float4*)(ap + 16);
            float4 f3 = *(const float4*)(ap + 20);
            float fa[16] = {f0.x, f0.y, f0.z, f0.w, f1.x, f1.y, f1.z, f1.w,
                            f2.x, f2.y, f2.z, f2.w, f3.x, f3.y, f3.z, f3.w};
#pragma unroll
            for (int e = 0; e < 16; ++e) {
                _Float16 hi, lo;
                split1(fa[e], hi, lo);
                ahi[e] = hi;
                alo[e] = lo;
            }
        }

#pragma unroll
        for (int i = 0; i < TILES_PER_WAVE; ++i) {
            const int tg = wave * TILES_PER_WAVE + i;          // 0..47
            const unsigned char* bp =
                wsW + (size_t)(tg * 8 + c) * 2048 + lane * 16;
            v8h bh0 = *(const v8h*)(bp);
            v8h bh1 = *(const v8h*)(bp + 512);
            v8h bl0 = *(const v8h*)(bp + 1024);
            v8h bl1 = *(const v8h*)(bp + 1536);
            v16h bhi, blo;
#pragma unroll
            for (int e = 0; e < 8; ++e) {
                bhi[e] = bh0[e]; bhi[e + 8] = bh1[e];
                blo[e] = bl0[e]; blo[e + 8] = bl1[e];
            }
            // hi*hi + lo*hi + hi*lo  (drop lo*lo): ~fp32-accurate product
            acc[i] = __builtin_amdgcn_wmma_f32_16x16x32_f16(
                false, ahi, false, bhi, (short)0, acc[i], false, false);
            acc[i] = __builtin_amdgcn_wmma_f32_16x16x32_f16(
                false, alo, false, bhi, (short)0, acc[i], false, false);
            acc[i] = __builtin_amdgcn_wmma_f32_16x16x32_f16(
                false, ahi, false, blo, (short)0, acc[i], false, false);
        }
    }

    // ---- Spill accumulators (+bias) to LDS in node-row order ---------------
    {
        const int ncol_lane = lane & 15;           // C/D layout: N = lane%16
        const int rbase = (lane & 16) ? 8 : 0;     // rows 0-7 / 8-15
#pragma unroll
        for (int i = 0; i < TILES_PER_WAVE; ++i) {
            const int tg  = wave * TILES_PER_WAVE + i;
            const int col = tg * 16 + ncol_lane;   // 0..767 (q|k|v cat)
            const float bc = wsB[col];
#pragma unroll
            for (int r = 0; r < 8; ++r)
                sQKV[(rbase + r) * QKV_STRIDE + col] = acc[i][r] + bc;
        }
    }
    __syncthreads();

    // ---- Attention phase: 8 threads per node, 16 nodes per workgroup -------
    const int node = tid >> 3;
    const int j    = tid & 7;
    const float* qr = sQKV + node * QKV_STRIDE;
    const float* kr = qr + 256;
    const float* vr = qr + 512;

    // each thread computes 2 of the 16 logits S[h][g] = (q_h . k_g) / 8
#pragma unroll
    for (int ee = 0; ee < 2; ++ee) {
        const int e  = j * 2 + ee;
        const int hh = e >> 2;
        const int g  = e & 3;
        const float4* q4 = (const float4*)(qr + hh * HEAD);
        const float4* k4 = (const float4*)(kr + g * HEAD);
        float s = 0.f;
#pragma unroll
        for (int d = 0; d < HEAD / 4; ++d) {
            float4 a = q4[d], b = k4[d];
            s += a.x * b.x + a.y * b.y + a.z * b.z + a.w * b.w;
        }
        sS[node * 16 + e] = s * 0.125f;            // 1/sqrt(64)
    }
    __syncthreads();

    // softmax over the 4-entry row + out = S' . v, 32 outputs per thread
    {
        const int hh = j >> 1;
        const float s0 = sS[node * 16 + hh * 4 + 0];
        const float s1 = sS[node * 16 + hh * 4 + 1];
        const float s2 = sS[node * 16 + hh * 4 + 2];
        const float s3 = sS[node * 16 + hh * 4 + 3];
        const float mx = fmaxf(fmaxf(s0, s1), fmaxf(s2, s3));
        float w0 = __expf(s0 - mx);
        float w1 = __expf(s1 - mx);
        float w2 = __expf(s2 - mx);
        float w3 = __expf(s3 - mx);
        const float inv = 1.f / (w0 + w1 + w2 + w3);
        w0 *= inv; w1 *= inv; w2 *= inv; w3 *= inv;

        const int row = m0 + node;
        if (row < n) {
            const int dbase = (j & 1) * 32;
            float* op = out + (size_t)row * OUT_SIZE + hh * HEAD + dbase;
            const float4* v0 = (const float4*)(vr + 0 * HEAD + dbase);
            const float4* v1 = (const float4*)(vr + 1 * HEAD + dbase);
            const float4* v2 = (const float4*)(vr + 2 * HEAD + dbase);
            const float4* v3 = (const float4*)(vr + 3 * HEAD + dbase);
#pragma unroll
            for (int d4 = 0; d4 < 8; ++d4) {
                float4 a = v0[d4], b = v1[d4], cc = v2[d4], dd = v3[d4];
                float4 o;
                o.x = w0 * a.x + w1 * b.x + w2 * cc.x + w3 * dd.x;
                o.y = w0 * a.y + w1 * b.y + w2 * cc.y + w3 * dd.y;
                o.z = w0 * a.z + w1 * b.z + w2 * cc.z + w3 * dd.z;
                o.w = w0 * a.w + w1 * b.w + w2 * cc.w + w3 * dd.w;
                *(float4*)(op + d4 * 4) = o;
            }
        }
    }
}

// ---------------------------------------------------------------------------
extern "C" void kernel_launch(void* const* d_in, const int* in_sizes, int n_in,
                              void* d_out, int out_size, void* d_ws, size_t ws_size,
                              hipStream_t stream) {
    const float* h  = (const float*)d_in[0];
    const float* Wk = (const float*)d_in[1];
    const float* bk = (const float*)d_in[2];
    const float* Wq = (const float*)d_in[3];
    const float* bq = (const float*)d_in[4];
    const float* Wv = (const float*)d_in[5];
    const float* bv = (const float*)d_in[6];
    float* out = (float*)d_out;

    const int n = in_sizes[0] / IN_SIZE;
    if (n <= 0) return;

    unsigned char* wsW = (unsigned char*)d_ws;
    float* wsB = (float*)((unsigned char*)d_ws + WS_WBYTES);

    // 196608 packed weight elements / 256 threads = 768 blocks
    pack_weights_kernel<<<768, 256, 0, stream>>>(Wq, Wk, Wv, bq, bk, bv, wsW, wsB);

    const int ntiles = (n + 15) / 16;
    qkv_attn_kernel<<<ntiles, 128, 0, stream>>>(h, wsW, wsB, out, n);
}
